// SpectralGraphConv_66597762892575
// MI455X (gfx1250) — matrix operand order repaired and checked by reading, not compiled, not run on hardware
//
#include <hip/hip_runtime.h>
#include <hip/hip_bf16.h>
#include <float.h>

#define BB  4
#define CC  64
#define NN  8192
#define KK  16
#define DD  64
#define KP1 17   // K filters + one slot holding -sum_k(W_k) for the self term

typedef __attribute__((ext_vector_type(16))) __bf16 v16bf;
typedef __attribute__((ext_vector_type(8)))  float  v8f;

struct __attribute__((aligned(16))) U128 { unsigned int x, y, z, w; };
union Frag { U128 q[2]; v16bf v; };   // 16 bf16 = 32 bytes = 8 VGPRs

__device__ __forceinline__ unsigned short f2bf(float f) {
  unsigned int u = __float_as_uint(f);
  u += 0x7fffu + ((u >> 16) & 1u);           // round-to-nearest-even
  return (unsigned short)(u >> 16);
}
__device__ __forceinline__ float bf2f(unsigned int h) {
  return __uint_as_float(h << 16);
}
__device__ __forceinline__ unsigned int pack2bf(float lo, float hi) {
  return (unsigned int)f2bf(lo) | ((unsigned int)f2bf(hi) << 16);
}

// ---------------------------------------------------------------------------
// 1) Pack x (B,C,N) f32 -> xT[b][n][c] bf16 (rows contiguous in c for WMMA A).
//    Converts in registers, writes 8 x 16B stores per thread.
// ---------------------------------------------------------------------------
__global__ void prep_x_kernel(const float* __restrict__ x,
                              unsigned short* __restrict__ xT) {
  int t = blockIdx.x * blockDim.x + threadIdx.x;   // B*N threads
  int b = t / NN, n = t % NN;
  const float* xp = x + (size_t)b * CC * NN + n;
  U128* o = (U128*)(xT + ((size_t)b * NN + n) * CC);
#pragma unroll
  for (int c8 = 0; c8 < 8; ++c8) {
    float v0 = xp[(size_t)(c8 * 8 + 0) * NN], v1 = xp[(size_t)(c8 * 8 + 1) * NN];
    float v2 = xp[(size_t)(c8 * 8 + 2) * NN], v3 = xp[(size_t)(c8 * 8 + 3) * NN];
    float v4 = xp[(size_t)(c8 * 8 + 4) * NN], v5 = xp[(size_t)(c8 * 8 + 5) * NN];
    float v6 = xp[(size_t)(c8 * 8 + 6) * NN], v7 = xp[(size_t)(c8 * 8 + 7) * NN];
    U128 r;
    r.x = pack2bf(v0, v1); r.y = pack2bf(v2, v3);
    r.z = pack2bf(v4, v5); r.w = pack2bf(v6, v7);
    o[c8] = r;
  }
}

// ---------------------------------------------------------------------------
// 2) Pack W (K,C,D) f32 -> Wt[k][d][c] bf16; slot k=16 holds -sum_k W[k][c][d]
// ---------------------------------------------------------------------------
__global__ void prep_w_kernel(const float* __restrict__ W,
                              unsigned short* __restrict__ Wt) {
  int t = blockIdx.x * blockDim.x + threadIdx.x;   // D*C threads
  int d = t >> 6, c = t & 63;
  float s = 0.f;
#pragma unroll
  for (int k = 0; k < KK; ++k) {
    float w = W[((size_t)k * CC + c) * DD + d];
    Wt[((size_t)k * DD + d) * CC + c] = f2bf(w);
    s += w;
  }
  Wt[((size_t)KK * DD + d) * CC + c] = f2bf(-s);
}

// ---------------------------------------------------------------------------
// 3) KNN: per query point keep 16 smallest squared distances (self included),
//    candidates streamed through LDS tiles; ascending-index strict-< insert
//    reproduces top_k's lower-index-first tie-break.
// ---------------------------------------------------------------------------
__global__ void knn_kernel(const float* __restrict__ pos, int* __restrict__ idx) {
  __shared__ float sx[128], sy[128], sz[128];
  const int tid = threadIdx.x;
  const int b = blockIdx.x / (NN / 128);
  const int n = (blockIdx.x % (NN / 128)) * 128 + tid;
  const float* px = pos + (size_t)b * 3 * NN;
  const float qx = px[n], qy = px[NN + n], qz = px[2 * NN + n];
  float bd[16]; int bi[16];
#pragma unroll
  for (int s = 0; s < 16; ++s) { bd[s] = FLT_MAX; bi[s] = 0; }
  for (int m0 = 0; m0 < NN; m0 += 128) {
    __syncthreads();
    sx[tid] = px[m0 + tid];
    sy[tid] = px[NN + m0 + tid];
    sz[tid] = px[2 * NN + m0 + tid];
    __syncthreads();
#pragma unroll 4
    for (int j = 0; j < 128; ++j) {
      float dx = sx[j] - qx, dy = sy[j] - qy, dz = sz[j] - qz;
      float d = __fmaf_rn(dx, dx, __fmaf_rn(dy, dy, dz * dz));
      if (d < bd[15]) {                      // rare: ~100 of 8192 candidates
        bd[15] = d; bi[15] = m0 + j;
#pragma unroll
        for (int s = 15; s > 0; --s) {
          if (bd[s] < bd[s - 1]) {
            float td = bd[s]; bd[s] = bd[s - 1]; bd[s - 1] = td;
            int   ti = bi[s]; bi[s] = bi[s - 1]; bi[s - 1] = ti;
          }
        }
      }
    }
  }
  int* op = idx + ((size_t)b * NN + n) * KK;
#pragma unroll
  for (int s = 0; s < 16; ++s) op[s] = bi[s];
}

// ---------------------------------------------------------------------------
// 4) Projection GEMMs: Y[b][k][n][d] = sum_c xT[b][n][c] * Wt[k][d][c]
//    128 threads = 4 waves; wave does 4 row-tiles of 16 x full D=64.
//    v_wmma_f32_16x16x32_bf16, K-loop = 2 (C=64).
//    YF32: Y stored f32 (fits MI455X 192MB L2) -> no pack VALU on store.
// ---------------------------------------------------------------------------
template <bool YF32>
__global__ void __launch_bounds__(128)
gemm_kernel(const unsigned short* __restrict__ xT,
            const unsigned short* __restrict__ Wt,
            void* __restrict__ Yv) {
  const int lane = threadIdx.x & 31;
  const int wave = threadIdx.x >> 5;
  const int k = blockIdx.y;          // 0..16
  const int b = blockIdx.z;
  const int hi  = (lane >= 16) ? 1 : 0;
  const int l15 = lane & 15;

  // B fragments (Wt rows are contiguous in c): held for all row-tiles.
  // B 32x16, contiguous k order used here (A matches): lanes 0-15 -> K=0..15 (N=lane), lanes 16-31 -> K=16..31.
  Frag bf[2][4];
#pragma unroll
  for (int ci = 0; ci < 2; ++ci) {
#pragma unroll
    for (int dt = 0; dt < 4; ++dt) {
      int d = dt * 16 + l15;
      int c = ci * 32 + hi * 16;
      const unsigned short* p = Wt + ((size_t)(k * DD + d)) * CC + c;
      bf[ci][dt].q[0] = *(const U128*)(p);
      bf[ci][dt].q[1] = *(const U128*)(p + 8);
    }
  }

  const size_t ybase = ((size_t)(b * KP1 + k)) * NN * DD;

#pragma unroll 1
  for (int rt = 0; rt < 4; ++rt) {
    const int row0 = blockIdx.x * 256 + wave * 64 + rt * 16;
    // A 16x32, contiguous k order used here (same as B): lane<16 holds K=0..15, lane>=16 K=16..31
    Frag af[2];
#pragma unroll
    for (int ci = 0; ci < 2; ++ci) {
      int row = row0 + l15;
      int c = ci * 32 + hi * 16;
      const unsigned short* p = xT + ((size_t)b * NN + row) * CC + c;
      af[ci].q[0] = *(const U128*)(p);
      af[ci].q[1] = *(const U128*)(p + 8);
    }
    v8f acc[4];
#pragma unroll
    for (int dt = 0; dt < 4; ++dt)
#pragma unroll
      for (int e = 0; e < 8; ++e) acc[dt][e] = 0.0f;

#pragma unroll
    for (int ci = 0; ci < 2; ++ci)
#pragma unroll
      for (int dt = 0; dt < 4; ++dt)
        acc[dt] = __builtin_amdgcn_wmma_f32_16x16x32_bf16(
            false, af[ci].v, false, bf[ci][dt].v, (short)0, acc[dt], false, false);

    // C/D layout: VGPR r -> (M=r, N=lane) lanes 0-15, (M=r+8, N=lane-16) lanes 16-31
#pragma unroll
    for (int dt = 0; dt < 4; ++dt) {
      int d = dt * 16 + l15;
#pragma unroll
      for (int r = 0; r < 8; ++r) {
        int row = row0 + r + hi * 8;
        if constexpr (YF32) {
          ((float*)Yv)[ybase + (size_t)row * DD + d] = acc[dt][r];
        } else {
          ((unsigned short*)Yv)[ybase + (size_t)row * DD + d] = f2bf(acc[dt][r]);
        }
      }
    }
  }
}

// ---------------------------------------------------------------------------
// 5) Gather + combine: out[b][d][n] = sum_k Y[b][k][idx[b,n,k]][:] + Y[b][16][n][:]
//    (slot 16 already carries the negated self projection). Rows are contiguous
//    (256B f32 / 128B bf16) and Y is L2-resident; prefetch next neighbor row.
// ---------------------------------------------------------------------------
template <bool YF32>
__global__ void gather_kernel(const void* __restrict__ Yv,
                              const int* __restrict__ idx,
                              float* __restrict__ out) {
  int t = blockIdx.x * blockDim.x + threadIdx.x;   // B*N threads
  int b = t / NN, n = t % NN;
  float acc[DD];

  int ids[KP1];
  const int* ip = idx + ((size_t)b * NN + n) * KK;
#pragma unroll
  for (int k = 0; k < KK; ++k) ids[k] = ip[k];
  ids[KK] = n;                                     // self term (slot 16 = -Wsum proj)

  if constexpr (YF32) {
    const float* yb = (const float*)Yv + ((size_t)(b * KP1)) * NN * DD;
    {
      const float4* p = (const float4*)(yb + ((size_t)KK * NN + n) * DD);
#pragma unroll
      for (int q = 0; q < 16; ++q) {
        float4 v = p[q];
        acc[q * 4 + 0] = v.x; acc[q * 4 + 1] = v.y;
        acc[q * 4 + 2] = v.z; acc[q * 4 + 3] = v.w;
      }
    }
#pragma unroll 1
    for (int k = 0; k < KK; ++k) {
      const float4* p = (const float4*)(yb + ((size_t)k * NN + ids[k]) * DD);
      __builtin_prefetch(yb + ((size_t)(k + 1) * NN + ids[k + 1]) * DD, 0, 0);
#pragma unroll
      for (int q = 0; q < 16; ++q) {
        float4 v = p[q];
        acc[q * 4 + 0] += v.x; acc[q * 4 + 1] += v.y;
        acc[q * 4 + 2] += v.z; acc[q * 4 + 3] += v.w;
      }
    }
  } else {
    const unsigned short* yb = (const unsigned short*)Yv + ((size_t)(b * KP1)) * NN * DD;
    {
      const U128* p = (const U128*)(yb + ((size_t)KK * NN + n) * DD);
#pragma unroll
      for (int q = 0; q < 8; ++q) {
        U128 v = p[q];
        acc[q * 8 + 0] = bf2f(v.x & 0xffffu); acc[q * 8 + 1] = bf2f(v.x >> 16);
        acc[q * 8 + 2] = bf2f(v.y & 0xffffu); acc[q * 8 + 3] = bf2f(v.y >> 16);
        acc[q * 8 + 4] = bf2f(v.z & 0xffffu); acc[q * 8 + 5] = bf2f(v.z >> 16);
        acc[q * 8 + 6] = bf2f(v.w & 0xffffu); acc[q * 8 + 7] = bf2f(v.w >> 16);
      }
    }
#pragma unroll 1
    for (int k = 0; k < KK; ++k) {
      const U128* p = (const U128*)(yb + ((size_t)k * NN + ids[k]) * DD);
      __builtin_prefetch(yb + ((size_t)(k + 1) * NN + ids[k + 1]) * DD, 0, 0);
#pragma unroll
      for (int q = 0; q < 8; ++q) {
        U128 v = p[q];
        acc[q * 8 + 0] += bf2f(v.x & 0xffffu); acc[q * 8 + 1] += bf2f(v.x >> 16);
        acc[q * 8 + 2] += bf2f(v.y & 0xffffu); acc[q * 8 + 3] += bf2f(v.y >> 16);
        acc[q * 8 + 4] += bf2f(v.z & 0xffffu); acc[q * 8 + 5] += bf2f(v.z >> 16);
        acc[q * 8 + 6] += bf2f(v.w & 0xffffu); acc[q * 8 + 7] += bf2f(v.w >> 16);
      }
    }
  }
  float* o = out + (size_t)b * DD * NN + n;        // output layout (B, D, N)
#pragma unroll
  for (int d = 0; d < DD; ++d) o[(size_t)d * NN] = acc[d];
}

// ---------------------------------------------------------------------------
// 6) BatchNorm (training stats over B*N per channel d), in place on d_out.
//    One block per channel; deterministic LDS tree reduction.
// ---------------------------------------------------------------------------
__global__ void bn_kernel(float* __restrict__ out,
                          const float* __restrict__ gamma,
                          const float* __restrict__ beta) {
  __shared__ float ssum[256], ssq[256];
  __shared__ float sc, sh;
  const int d = blockIdx.x, tid = threadIdx.x;
  float sum = 0.f, sq = 0.f;
  for (int b = 0; b < BB; ++b) {
    const float* p = out + ((size_t)b * DD + d) * NN;
    for (int n = tid; n < NN; n += 256) { float v = p[n]; sum += v; sq += v * v; }
  }
  ssum[tid] = sum; ssq[tid] = sq;
  __syncthreads();
  for (int s = 128; s > 0; s >>= 1) {
    if (tid < s) { ssum[tid] += ssum[tid + s]; ssq[tid] += ssq[tid + s]; }
    __syncthreads();
  }
  if (tid == 0) {
    const float inv = 1.0f / (float)(BB * NN);
    float mean = ssum[0] * inv;
    float var  = ssq[0] * inv - mean * mean;
    float scale = rsqrtf(var + 1e-5f) * gamma[d];
    sc = scale; sh = beta[d] - mean * scale;
  }
  __syncthreads();
  float scale = sc, shift = sh;
  for (int b = 0; b < BB; ++b) {
    float* p = out + ((size_t)b * DD + d) * NN;
    for (int n = tid; n < NN; n += 256) p[n] = p[n] * scale + shift;
  }
}

// ---------------------------------------------------------------------------
extern "C" void kernel_launch(void* const* d_in, const int* in_sizes, int n_in,
                              void* d_out, int out_size, void* d_ws, size_t ws_size,
                              hipStream_t stream) {
  const float* x     = (const float*)d_in[0];   // (B, C, N)
  const float* pos   = (const float*)d_in[1];   // (B, 3, N)
  const float* W     = (const float*)d_in[2];   // (K, C, D)
  const float* gamma = (const float*)d_in[3];   // (D,)
  const float* beta  = (const float*)d_in[4];   // (D,)
  float* out = (float*)d_out;                   // (B, D, N)
  char*  ws  = (char*)d_ws;

  const size_t IDX_OFF = 0;                                                  // 2 MiB
  const size_t XT_OFF  = IDX_OFF + (size_t)BB * NN * KK * sizeof(int);
  const size_t WT_OFF  = XT_OFF + (size_t)BB * NN * CC * sizeof(unsigned short);
  const size_t Y_OFF   = (WT_OFF + (size_t)KP1 * DD * CC * sizeof(unsigned short)
                          + 255) & ~(size_t)255;
  const size_t Y_ELEMS = (size_t)BB * KP1 * NN * DD;

  int*            idx = (int*)(ws + IDX_OFF);
  unsigned short* xT  = (unsigned short*)(ws + XT_OFF);
  unsigned short* Wt  = (unsigned short*)(ws + WT_OFF);
  void*           Y   = (void*)(ws + Y_OFF);

  // Prefer f32 Y (142 MB: fits MI455X's 192 MB L2, kills pack/unpack VALU);
  // fall back to bf16 Y (71 MB) if scratch is tight. ws_size is fixed across
  // calls, so the choice is deterministic.
  const bool yf32 = (ws_size >= Y_OFF + Y_ELEMS * sizeof(float));

  prep_x_kernel<<<BB * NN / 256, 256, 0, stream>>>(x, xT);
  prep_w_kernel<<<DD * CC / 256, 256, 0, stream>>>(W, Wt);
  knn_kernel<<<BB * NN / 128, 128, 0, stream>>>(pos, idx);
  if (yf32) {
    gemm_kernel<true><<<dim3(NN / 256, KP1, BB), 128, 0, stream>>>(xT, Wt, Y);
    gather_kernel<true><<<BB * NN / 128, 128, 0, stream>>>(Y, idx, out);
  } else {
    gemm_kernel<false><<<dim3(NN / 256, KP1, BB), 128, 0, stream>>>(xT, Wt, Y);
    gather_kernel<false><<<BB * NN / 128, 128, 0, stream>>>(Y, idx, out);
  }
  bn_kernel<<<DD, 256, 0, stream>>>(out, gamma, beta);
}
